// CapsuleNet_30588757082752
// MI455X (gfx1250) — matrix operand order repaired and checked by reading, not compiled, not run on hardware
//
#include <hip/hip_runtime.h>
#include <hip/hip_bf16.h>
#include <math.h>

typedef __attribute__((ext_vector_type(16))) _Float16     v16h;
typedef __attribute__((ext_vector_type(8)))  _Float16     v8h;
typedef __attribute__((ext_vector_type(8)))  float        v8f;
typedef __attribute__((ext_vector_type(4)))  unsigned int u32x4;
typedef __attribute__((ext_vector_type(4)))  int          i32x4;
typedef __attribute__((ext_vector_type(8)))  int          i32x8;

// ---------------- problem constants ----------------
constexpr int BATCH = 128, CIN = 30, HW = 15, CLASSES = 16;
constexpr int H1 = 13, NCOL1 = BATCH * H1 * H1;       // 21632
constexpr int K1 = CIN * 9, K1P = 288;                // 270 -> pad 288
constexpr int M1 = 256;
constexpr int H2 = 11, NCOL2 = BATCH * H2 * H2;       // 15488
constexpr int K2 = 256 * 9;                           // 2304 (already /32)
constexpr int M2 = 256;
constexpr int NCAPS = 3872, IND = 8, OUTD = 16;
constexpr int DDIM = 256, F1 = 328, F1P = 352, F2 = 192, F3 = CIN * HW * HW; // 6750

#if __has_builtin(__builtin_amdgcn_tensor_load_to_lds)
#define HAVE_TDM 1
#endif

#ifdef HAVE_TDM
// 2D-tile D# (ISA 8.3/8.4): dim0 = contiguous K (32 elems, 64B), dim1 = N rows.
// tensor dims are relative to the tile origin so OOB rows zero-fill the LDS tile.
// LDS padding: 4 DWORDs (16B) after every 16 DWORDs (64B) -> 80B row pitch.
static __device__ __forceinline__ void tdm_load_2d(
    unsigned lds_byte_addr, const _Float16* gptr,
    unsigned tensor_d0, unsigned tensor_d1,
    unsigned tile_d0, unsigned tile_d1,
    unsigned long long stride0_elems)
{
  unsigned long long ga = (unsigned long long)(uintptr_t)gptr;
  u32x4 g0 = {
    1u,                                               // count=1, user mode
    lds_byte_addr,                                    // LDS byte address
    (unsigned)(ga & 0xFFFFFFFFu),                     // global addr lo
    (unsigned)((ga >> 32) & 0x1FFFFFFu) | (2u << 30)  // addr hi + type=2
  };
  // w0: data_size=1 (2B) | pad_enable | pad_interval=3 (16 dw) | pad_amount=3 (4 dw)
  i32x8 g1 = {
    (int)((1u << 16) | (1u << 20) | (3u << 22) | (3u << 25)),
    (int)((tensor_d0 & 0xFFFFu) << 16),
    (int)(((tensor_d0 >> 16) & 0xFFFFu) | ((tensor_d1 & 0xFFFFu) << 16)),
    (int)(((tensor_d1 >> 16) & 0xFFFFu) | ((tile_d0 & 0xFFFFu) << 16)),
    (int)(tile_d1 & 0xFFFFu),                         // tile_dim1, tile_dim2=0
    (int)(stride0_elems & 0xFFFFFFFFull),
    (int)((stride0_elems >> 32) & 0xFFFFull),         // stride0 hi, stride1=0
    0
  };
  i32x4 z4 = {0, 0, 0, 0};
#if __clang_major__ >= 23
  i32x8 z8 = {0, 0, 0, 0, 0, 0, 0, 0};
  __builtin_amdgcn_tensor_load_to_lds(g0, g1, z4, z4, z8, 0);
#else
  __builtin_amdgcn_tensor_load_to_lds(g0, g1, z4, z4, 0);
#endif
}
#endif

// ---------------- generic WMMA GEMM ----------------
// C[M,N](f32) = A[M,Kpad](f16, zero-padded) * Bt[N,Kpad](f16, B transposed, padded)
// K passed = Kpad (multiple of 32). M must be a multiple of 64 (here 128/256).
// 256 threads = 8 waves; block tile 64(M) x 64(N); wave = 16(M) x 32(N) -> 2 WMMAs/step.
// B panels are double-buffered in LDS via the Tensor Data Mover: wave 0 queues the
// next panel, waits tensorcnt<=1 (current panel landed, next in flight), and a
// barrier releases compute overlapped with the next DMA (TENSORcnt is in-order).
// biasMode: 0=none 1=per-M 2=per-N ; act: 0=none 1=relu 2=sigmoid
// Ch (optional f16 out) has padded leading dim ldch; pad columns written as 0.
__global__ void wmma_gemm(const _Float16* __restrict__ A,
                          const _Float16* __restrict__ Bt,
                          const float* __restrict__ bias,
                          float* __restrict__ C,
                          _Float16* __restrict__ Ch,
                          int M, int N, int K, int lda, int ldbt, int ldch,
                          int act, int biasMode)
{
  constexpr int TP = 40;                      // LDS row pitch in f16 (64B data + 16B pad)
  __shared__ __align__(128) _Float16 tile[2][64 * TP];   // 2 x 5120 B
  const int tid  = threadIdx.x;
  const int lane = tid & 31;
  const int wave = tid >> 5;
  const int m0     = blockIdx.y * 64 + (wave >> 1) * 16;
  const int nBase  = (wave & 1) * 32;
  const int nPanel = blockIdx.x * 64;
  const int nl = lane & 15;
  const int kh = lane >> 4;

  v8f acc[2];
  acc[0] = (v8f){0.f,0.f,0.f,0.f,0.f,0.f,0.f,0.f};
  acc[1] = (v8f){0.f,0.f,0.f,0.f,0.f,0.f,0.f,0.f};

  const _Float16* aRow = A + (size_t)(m0 + nl) * lda + kh * 16;

#ifdef HAVE_TDM
  if (tid < 32) {                             // prologue: queue first panel
    tdm_load_2d((unsigned)(uintptr_t)(&tile[0][0]),
                Bt + (size_t)nPanel * ldbt,
                (unsigned)ldbt, (unsigned)(N - nPanel),
                32u, 64u, (unsigned long long)ldbt);
  }
#endif

  int cur = 0;
  for (int k0 = 0; k0 < K; k0 += 32, cur ^= 1) {
#ifdef HAVE_TDM
    if (tid < 32) {
      int kn = k0 + 32;
      if (kn < K) {                           // queue next panel into other buffer
        tdm_load_2d((unsigned)(uintptr_t)(&tile[cur ^ 1][0]),
                    Bt + (size_t)nPanel * ldbt + kn,
                    (unsigned)(ldbt - kn), (unsigned)(N - nPanel),
                    32u, 64u, (unsigned long long)ldbt);
        __builtin_amdgcn_s_wait_tensorcnt(1); // current done, next may be in flight
      } else {
        __builtin_amdgcn_s_wait_tensorcnt(0);
      }
    }
#else
    for (int i = tid; i < 64 * 32; i += 256) {
      int r = i >> 5, c = i & 31;
      int gn = nPanel + r;
      tile[cur][r * TP + c] = (gn < N) ? Bt[(size_t)gn * ldbt + k0 + c] : (_Float16)0.f;
    }
#endif
    __syncthreads();                          // staged panel visible to all 8 waves

    // A fragment: k = kh*16 + e contiguous per lane -> two contiguous 16B loads
    v8h a_lo = *(const v8h*)(aRow + k0);
    v8h a_hi = *(const v8h*)(aRow + k0 + 8);
    v16h af = __builtin_shufflevector(a_lo, a_hi,
                                      0, 1, 2, 3, 4, 5, 6, 7,
                                      8, 9, 10, 11, 12, 13, 14, 15);
    // B fragments from transposed LDS tile: k = kh*16 + e contiguous per lane
    const _Float16* br0 = &tile[cur][(nBase + nl) * TP + kh * 16];
    const _Float16* br1 = br0 + 16 * TP;
    v16h bf0 = __builtin_shufflevector(*(const v8h*)br0, *(const v8h*)(br0 + 8),
                                       0, 1, 2, 3, 4, 5, 6, 7,
                                       8, 9, 10, 11, 12, 13, 14, 15);
    v16h bf1 = __builtin_shufflevector(*(const v8h*)br1, *(const v8h*)(br1 + 8),
                                       0, 1, 2, 3, 4, 5, 6, 7,
                                       8, 9, 10, 11, 12, 13, 14, 15);
    acc[0] = __builtin_amdgcn_wmma_f32_16x16x32_f16(
        false, af, false, bf0, (short)0, acc[0], false, false);
    acc[1] = __builtin_amdgcn_wmma_f32_16x16x32_f16(
        false, af, false, bf1, (short)0, acc[1], false, false);

    __syncthreads();                          // reads done before buffer is re-filled
  }

  // C/D layout: n = lane&15, m = m0 + (lane>>4)*8 + r
  #pragma unroll
  for (int sub = 0; sub < 2; ++sub) {
    int n = nPanel + nBase + sub * 16 + nl;
    #pragma unroll
    for (int r = 0; r < 8; ++r) {
      int m = m0 + (kh << 3) + r;
      float v = acc[sub][r];
      if (biasMode == 1)                 v += bias[m];
      else if (biasMode == 2 && n < N)   v += bias[n];
      if (act == 1)      v = fmaxf(v, 0.f);
      else if (act == 2) v = 1.f / (1.f + __expf(-v));
      if (C && n < N)     C[(size_t)m * N + n] = v;
      if (Ch && n < ldch) Ch[(size_t)m * ldch + n] = (n < N) ? (_Float16)v : (_Float16)0.f;
    }
  }
}

// ---------------- helper kernels ----------------
__global__ void fill_zero(float* p, long long n) {
  long long t = (long long)blockIdx.x * blockDim.x + threadIdx.x;
  for (; t < n; t += (long long)gridDim.x * blockDim.x) p[t] = 0.f;
}

// A-weight pack: [M,K] f32 -> [M,Kpad] f16 zero-padded
__global__ void cvt_pad(const float* __restrict__ s, _Float16* __restrict__ d,
                        int M, int K, int Kpad) {
  long long t = (long long)blockIdx.x * blockDim.x + threadIdx.x;
  const long long total = (long long)M * Kpad;
  for (; t < total; t += (long long)gridDim.x * blockDim.x) {
    int m = (int)(t / Kpad), k = (int)(t % Kpad);
    d[t] = (k < K) ? (_Float16)s[(size_t)m * K + k] : (_Float16)0.f;
  }
}

// B-weight pack: [K,N] f32 -> transposed [N,Kpad] f16 zero-padded
__global__ void cvt_tpad(const float* __restrict__ s, _Float16* __restrict__ d,
                         int K, int N, int Kpad) {
  long long t = (long long)blockIdx.x * blockDim.x + threadIdx.x;
  const long long total = (long long)N * Kpad;
  for (; t < total; t += (long long)gridDim.x * blockDim.x) {
    int n = (int)(t / Kpad), k = (int)(t % Kpad);
    d[t] = (k < K) ? (_Float16)s[(size_t)k * N + n] : (_Float16)0.f;
  }
}

// conv1 im2col, transposed+padded: B1t[n, kpad] ; n = b*169+oh*13+ow ; k = c*9+kh*3+kw
__global__ void im2col_conv1_t(const float* __restrict__ x, _Float16* __restrict__ B1t) {
  long long t = (long long)blockIdx.x * blockDim.x + threadIdx.x;
  const long long total = (long long)NCOL1 * K1P;
  for (; t < total; t += (long long)gridDim.x * blockDim.x) {
    int n = (int)(t / K1P), k = (int)(t % K1P);
    _Float16 v = (_Float16)0.f;
    if (k < K1) {
      int c = k / 9, r = k % 9, kh = r / 3, kw = r % 3;
      int b = n / 169, s = n % 169, oh = s / 13, ow = s % 13;
      v = (_Float16)x[(((size_t)b * CIN + c) * HW + oh + kh) * HW + ow + kw];
    }
    B1t[t] = v;
  }
}

// per-channel batch stats over C1 rows -> scale/shift (biased variance)
__global__ void bn_stats(const float* __restrict__ C1,
                         const float* __restrict__ gamma,
                         const float* __restrict__ beta,
                         float* __restrict__ scl, float* __restrict__ shf) {
  __shared__ float s1[256], s2[256];
  int m = blockIdx.x, tid = threadIdx.x;
  float a = 0.f, b = 0.f;
  for (int n = tid; n < NCOL1; n += 256) {
    float v = C1[(size_t)m * NCOL1 + n];
    a += v; b += v * v;
  }
  s1[tid] = a; s2[tid] = b;
  __syncthreads();
  for (int off = 128; off > 0; off >>= 1) {
    if (tid < off) { s1[tid] += s1[tid + off]; s2[tid] += s2[tid + off]; }
    __syncthreads();
  }
  if (tid == 0) {
    float mean = s1[0] / (float)NCOL1;
    float var  = s2[0] / (float)NCOL1 - mean * mean;
    float sc   = gamma[m] * rsqrtf(var + 1e-5f);
    scl[m] = sc;
    shf[m] = beta[m] - mean * sc;
  }
}

// conv2 im2col fused with BN+ReLU, transposed: B2t[n2, k2] f16 (K2 already /32)
__global__ void build_B2_t(const float* __restrict__ C1,
                           const float* __restrict__ scl,
                           const float* __restrict__ shf,
                           _Float16* __restrict__ B2t) {
  long long t = (long long)blockIdx.x * blockDim.x + threadIdx.x;
  const long long total = (long long)NCOL2 * K2;
  for (; t < total; t += (long long)gridDim.x * blockDim.x) {
    int n = (int)(t / K2), k = (int)(t % K2);
    int c = k / 9, r = k % 9, kh = r / 3, kw = r % 3;
    int b = n / 121, s = n % 121, oh = s / 11, ow = s % 11;
    int col = b * 169 + (oh + kh) * 13 + (ow + kw);
    float v = C1[(size_t)c * NCOL1 + col];
    v = fmaxf(v * scl[c] + shf[c], 0.f);
    B2t[t] = (_Float16)v;
  }
}

// squash primary capsules: C2[256,15488] -> caps[b,n,j] f32
__global__ void squash_caps(const float* __restrict__ C2, float* __restrict__ caps) {
  int t = blockIdx.x * blockDim.x + threadIdx.x;
  if (t >= BATCH * NCAPS) return;
  int b = t / NCAPS, n = t % NCAPS;
  float v[IND], n2 = 0.f;
  #pragma unroll
  for (int j = 0; j < IND; ++j) {
    int f = n * IND + j, c8 = f / 121, s = f % 121;
    float x = C2[(size_t)c8 * NCOL2 + b * 121 + s];
    v[j] = x; n2 += x * x;
  }
  float nrm = sqrtf(n2);
  float sc = n2 / (1.f + n2) / (nrm + 1e-8f);
  #pragma unroll
  for (int j = 0; j < IND; ++j)
    caps[((size_t)b * NCAPS + n) * IND + j] = sc * v[j];
}

// softmax over the 16 classes (axis=1 of b_log[B,16,3872])
__global__ void softmax_classes(const float* __restrict__ blog, float* __restrict__ cbuf) {
  int t = blockIdx.x * blockDim.x + threadIdx.x;
  if (t >= BATCH * NCAPS) return;
  int b = t / NCAPS, n = t % NCAPS;
  float v[CLASSES], mx = -1e30f;
  #pragma unroll
  for (int o = 0; o < CLASSES; ++o) {
    v[o] = blog[((size_t)b * CLASSES + o) * NCAPS + n];
    mx = fmaxf(mx, v[o]);
  }
  float s = 0.f;
  #pragma unroll
  for (int o = 0; o < CLASSES; ++o) { v[o] = __expf(v[o] - mx); s += v[o]; }
  float inv = 1.f / s;
  #pragma unroll
  for (int o = 0; o < CLASSES; ++o)
    cbuf[((size_t)b * CLASSES + o) * NCAPS + n] = v[o] * inv;
}

// out[b,o,:] = squash( sum_n c[b,o,n] * x_hat[b,o,n,:] ), x_hat recomputed on the fly
__global__ void route_out(const float* __restrict__ cbuf,
                          const float* __restrict__ caps,
                          const float* __restrict__ capsW,
                          float* __restrict__ outv) {
  int b = blockIdx.x >> 4, o = blockIdx.x & 15, tid = threadIdx.x;
  float acc[OUTD];
  #pragma unroll
  for (int d = 0; d < OUTD; ++d) acc[d] = 0.f;

  for (int n = tid; n < NCAPS; n += 256) {
    float cj[IND];
    const float* cp = caps + ((size_t)b * NCAPS + n) * IND;
    #pragma unroll
    for (int j = 0; j < IND; ++j) cj[j] = cp[j];
    float cc = cbuf[((size_t)b * CLASSES + o) * NCAPS + n];
    const float* wp = capsW + ((size_t)o * NCAPS + n) * OUTD * IND;
    #pragma unroll
    for (int d = 0; d < OUTD; ++d) {
      float xh = 0.f;
      #pragma unroll
      for (int j = 0; j < IND; ++j) xh += wp[d * IND + j] * cj[j];
      acc[d] += cc * xh;
    }
  }

  __shared__ float red[OUTD * 256];
  #pragma unroll
  for (int d = 0; d < OUTD; ++d) red[d * 256 + tid] = acc[d];
  __syncthreads();
  if (tid < OUTD) {
    float s = 0.f;
    for (int i = 0; i < 256; ++i) s += red[tid * 256 + i];
    red[tid * 256] = s;
  }
  __syncthreads();
  if (tid == 0) {
    float n2 = 0.f;
    #pragma unroll
    for (int d = 0; d < OUTD; ++d) n2 += red[d * 256] * red[d * 256];
    float nrm = sqrtf(n2);
    float sc = n2 / (1.f + n2) / (nrm + 1e-8f);
    #pragma unroll
    for (int d = 0; d < OUTD; ++d)
      outv[((size_t)b * CLASSES + o) * OUTD + d] = sc * red[d * 256];
  }
}

// b_log[b,o,n] += dot(out[b,o,:], x_hat[b,o,n,:])  (x_hat recomputed)
__global__ void update_b(const float* __restrict__ outv,
                         const float* __restrict__ caps,
                         const float* __restrict__ capsW,
                         float* __restrict__ blog) {
  int b = blockIdx.x >> 4, o = blockIdx.x & 15, tid = threadIdx.x;
  __shared__ float ov[OUTD];
  if (tid < OUTD) ov[tid] = outv[((size_t)b * CLASSES + o) * OUTD + tid];
  __syncthreads();
  float od[OUTD];
  #pragma unroll
  for (int d = 0; d < OUTD; ++d) od[d] = ov[d];

  for (int n = tid; n < NCAPS; n += 256) {
    float cj[IND];
    const float* cp = caps + ((size_t)b * NCAPS + n) * IND;
    #pragma unroll
    for (int j = 0; j < IND; ++j) cj[j] = cp[j];
    const float* wp = capsW + ((size_t)o * NCAPS + n) * OUTD * IND;
    float delta = 0.f;
    #pragma unroll
    for (int d = 0; d < OUTD; ++d) {
      float xh = 0.f;
      #pragma unroll
      for (int j = 0; j < IND; ++j) xh += wp[d * IND + j] * cj[j];
      delta += od[d] * xh;
    }
    blog[((size_t)b * CLASSES + o) * NCAPS + n] += delta;
  }
}

// length to d_out[0:2048]; masked capsules (f16) for the decoder
__global__ void length_mask(const float* __restrict__ outv,
                            const float* __restrict__ y,
                            float* __restrict__ lenOut,
                            _Float16* __restrict__ maskedh) {
  int t = blockIdx.x * blockDim.x + threadIdx.x;
  if (t >= BATCH * CLASSES) return;
  int b = t / CLASSES, o = t % CLASSES;
  float n2 = 0.f, vv[OUTD];
  #pragma unroll
  for (int d = 0; d < OUTD; ++d) {
    float v = outv[(size_t)t * OUTD + d];
    vv[d] = v; n2 += v * v;
  }
  lenOut[t] = sqrtf(n2);
  float yv = y[t];
  #pragma unroll
  for (int d = 0; d < OUTD; ++d)
    maskedh[(size_t)b * DDIM + o * OUTD + d] = (_Float16)(vv[d] * yv);
}

// ---------------- host orchestration ----------------
extern "C" void kernel_launch(void* const* d_in, const int* in_sizes, int n_in,
                              void* d_out, int out_size, void* d_ws, size_t ws_size,
                              hipStream_t stream) {
  (void)in_sizes; (void)n_in; (void)out_size; (void)ws_size;
  const float* x     = (const float*)d_in[0];
  const float* y     = (const float*)d_in[1];
  const float* w1    = (const float*)d_in[2];
  const float* b1    = (const float*)d_in[3];
  const float* gamma = (const float*)d_in[4];
  const float* beta  = (const float*)d_in[5];
  const float* w2    = (const float*)d_in[6];
  const float* b2    = (const float*)d_in[7];
  const float* capsW = (const float*)d_in[8];
  const float* dw1   = (const float*)d_in[9];
  const float* db1   = (const float*)d_in[10];
  const float* dw2   = (const float*)d_in[11];
  const float* db2   = (const float*)d_in[12];
  const float* dw3   = (const float*)d_in[13];
  const float* db3   = (const float*)d_in[14];
  float* outp = (float*)d_out;

  char* p = (char*)d_ws;
  auto alloc = [&](size_t bytes) -> void* {
    void* r = (void*)p;
    p += (bytes + 255) & ~(size_t)255;
    return r;
  };
  _Float16* W1p   = (_Float16*)alloc((size_t)M1 * K1P * 2);
  _Float16* W2p   = (_Float16*)alloc((size_t)M2 * K2 * 2);
  _Float16* Wd1t  = (_Float16*)alloc((size_t)F1 * DDIM * 2);   // [328][256]
  _Float16* Wd2t  = (_Float16*)alloc((size_t)F2 * F1P * 2);    // [192][352]
  _Float16* Wd3t  = (_Float16*)alloc((size_t)F3 * F2 * 2);     // [6750][192]
  _Float16* B1t   = (_Float16*)alloc((size_t)NCOL1 * K1P * 2);
  float*    C1    = (float*)   alloc((size_t)M1 * NCOL1 * 4);
  float*    bnscl = (float*)   alloc(256 * 4);
  float*    bnshf = (float*)   alloc(256 * 4);
  _Float16* B2t   = (_Float16*)alloc((size_t)NCOL2 * K2 * 2);
  float*    C2    = (float*)   alloc((size_t)M2 * NCOL2 * 4);
  float*    caps  = (float*)   alloc((size_t)BATCH * NCAPS * IND * 4);
  float*    blog  = (float*)   alloc((size_t)BATCH * CLASSES * NCAPS * 4);
  float*    cbuf  = (float*)   alloc((size_t)BATCH * CLASSES * NCAPS * 4);
  float*    outv  = (float*)   alloc((size_t)BATCH * CLASSES * OUTD * 4);
  _Float16* mskh  = (_Float16*)alloc((size_t)BATCH * DDIM * 2);
  _Float16* r1h   = (_Float16*)alloc((size_t)BATCH * F1P * 2); // padded 328->352
  _Float16* r2h   = (_Float16*)alloc((size_t)BATCH * F2 * 2);

  auto blocks = [](long long n) { return (unsigned)((n + 255) / 256); };

  // weight packing (A padded, B transposed+padded)
  cvt_pad <<<blocks((long long)M1 * K1P), 256, 0, stream>>>(w1, W1p, M1, K1, K1P);
  cvt_pad <<<blocks((long long)M2 * K2),  256, 0, stream>>>(w2, W2p, M2, K2, K2);
  cvt_tpad<<<blocks((long long)F1 * DDIM), 256, 0, stream>>>(dw1, Wd1t, DDIM, F1, DDIM);
  cvt_tpad<<<blocks((long long)F2 * F1P),  256, 0, stream>>>(dw2, Wd2t, F1, F2, F1P);
  cvt_tpad<<<blocks((long long)F3 * F2),   256, 0, stream>>>(dw3, Wd3t, F2, F3, F2);

  // conv1 as GEMM
  im2col_conv1_t<<<blocks((long long)NCOL1 * K1P), 256, 0, stream>>>(x, B1t);
  wmma_gemm<<<dim3((NCOL1 + 63) / 64, M1 / 64), 256, 0, stream>>>(
      W1p, B1t, b1, C1, nullptr, M1, NCOL1, K1P, K1P, K1P, 0, /*act*/0, /*bias*/1);

  // batchnorm + relu fused into conv2 im2col
  bn_stats<<<256, 256, 0, stream>>>(C1, gamma, beta, bnscl, bnshf);
  build_B2_t<<<blocks((long long)NCOL2 * K2), 256, 0, stream>>>(C1, bnscl, bnshf, B2t);
  wmma_gemm<<<dim3((NCOL2 + 63) / 64, M2 / 64), 256, 0, stream>>>(
      W2p, B2t, b2, C2, nullptr, M2, NCOL2, K2, K2, K2, 0, /*act*/0, /*bias*/1);

  // primary capsules + dynamic routing (x_hat recomputed, never materialized)
  squash_caps<<<blocks(BATCH * NCAPS), 256, 0, stream>>>(C2, caps);
  fill_zero<<<1024, 256, 0, stream>>>(blog, (long long)BATCH * CLASSES * NCAPS);
  for (int it = 0; it < 3; ++it) {
    softmax_classes<<<blocks(BATCH * NCAPS), 256, 0, stream>>>(blog, cbuf);
    route_out<<<BATCH * CLASSES, 256, 0, stream>>>(cbuf, caps, capsW, outv);
    if (it < 2)
      update_b<<<BATCH * CLASSES, 256, 0, stream>>>(outv, caps, capsW, blog);
  }

  // lengths + masked decoder input
  length_mask<<<blocks(BATCH * CLASSES), 256, 0, stream>>>(outv, y, outp, mskh);

  // decoder MLP (sigmoid, sigmoid, linear+bias); K tails handled by zero padding
  wmma_gemm<<<dim3((F1 + 63) / 64, BATCH / 64), 256, 0, stream>>>(
      mskh, Wd1t, db1, nullptr, r1h, BATCH, F1, DDIM, DDIM, DDIM, F1P, /*act*/2, /*bias*/2);
  wmma_gemm<<<dim3((F2 + 63) / 64, BATCH / 64), 256, 0, stream>>>(
      r1h, Wd2t, db2, nullptr, r2h, BATCH, F2, F1P, F1P, F1P, F2, /*act*/2, /*bias*/2);
  wmma_gemm<<<dim3((F3 + 63) / 64, BATCH / 64), 256, 0, stream>>>(
      r2h, Wd3t, db3, outp + BATCH * CLASSES, nullptr, BATCH, F3, F2, F2, F2, 0,
      /*act*/0, /*bias*/2);
}
